// BinConv2d_50895362457819
// MI455X (gfx1250) — 8-bit matrix operand order repaired and checked by reading, compiled, not run on hardware
//
#include <hip/hip_runtime.h>
#include <cstdint>

typedef int   v8i __attribute__((ext_vector_type(8)));
typedef int   v4i __attribute__((ext_vector_type(4)));

#define Nn    32
#define CIN   256
#define COUT  256
#define Hh    56
#define Ww    56
#define HW    (Hh * Ww)
#define NPIX  (Nn * HW)
#define HP    58
#define WP    58
#define ROWB  (WP * 256)
#define EPSV  1e-4f

__global__ void k_bn_stats(const float* __restrict__ x,
                           const float* __restrict__ gamma,
                           const float* __restrict__ beta,
                           float* __restrict__ scale,
                           float* __restrict__ shift) {
  const int c   = blockIdx.x;
  const int tid = threadIdx.x;
  const float* xc = x + (size_t)c * HW;
  float s = 0.f, s2 = 0.f;
  for (int i = tid; i < NPIX; i += 256) {
    int n  = i / HW;
    int hw = i - n * HW;
    float v = xc[(size_t)n * (CIN * HW) + hw];
    s  += v;
    s2 += v * v;
  }
  __shared__ float ls[256], ls2[256];
  ls[tid] = s; ls2[tid] = s2;
  __syncthreads();
  for (int off = 128; off > 0; off >>= 1) {
    if (tid < off) { ls[tid] += ls[tid + off]; ls2[tid] += ls2[tid + off]; }
    __syncthreads();
  }
  if (tid == 0) {
    float mu  = ls[0]  * (1.0f / (float)NPIX);
    float var = ls2[0] * (1.0f / (float)NPIX) - mu * mu;
    float inv = rsqrtf(var + EPSV);
    float sc  = gamma[c] * inv;
    scale[c] = sc;
    shift[c] = beta[c] - mu * sc;
  }
}

__global__ void k_padzero(int8_t* __restrict__ xb) {
  const int per_n = 228 * 16;
  int idx = blockIdx.x * blockDim.x + threadIdx.x;
  if (idx >= Nn * per_n) return;
  int n  = idx / per_n;
  int r  = idx - n * per_n;
  int p  = r >> 4;
  int ch = r & 15;
  int h, w;
  if      (p < 58)  { h = 0;           w = p;       }
  else if (p < 116) { h = HP - 1;      w = p - 58;  }
  else if (p < 172) { h = p - 116 + 1; w = 0;       }
  else              { h = p - 172 + 1; w = WP - 1;  }
  v4i z = {};
  *(v4i*)(xb + ((size_t)(n * HP + h) * WP + w) * 256 + ch * 16) = z;
}

__global__ void k_binact(const float* __restrict__ x,
                         const float* __restrict__ scale,
                         const float* __restrict__ shift,
                         int8_t* __restrict__ xb,
                         float* __restrict__ m) {
  __shared__ int8_t sb[64 * 256];
  __shared__ float  psum[4][64];
  __shared__ float  ssc[256], ssh[256];

  const int tid = threadIdx.x;
  ssc[tid] = scale[tid];
  ssh[tid] = shift[tid];
  __syncthreads();

  const int pix = tid & 63;
  const int cq  = tid >> 6;
  const int p0  = blockIdx.x * 64;
  const int n   = p0 / HW;
  const int hw0 = p0 - n * HW;

  const float* xp = x + (size_t)n * CIN * HW + hw0 + pix;
  float asum = 0.f;
  #pragma unroll 4
  for (int j = 0; j < 64; ++j) {
    int c = cq * 64 + j;
    float v = xp[(size_t)c * HW];
    v = v * ssc[c] + ssh[c];
    asum += fabsf(v);
    int8_t sg = (v > 0.f) ? (int8_t)1 : ((v < 0.f) ? (int8_t)-1 : (int8_t)0);
    int jj = (((j >> 3) & 1) << 5) | (((j >> 4) & 3) << 3) | (j & 7);
    sb[pix * 256 + cq * 64 + jj] = sg;
  }
  psum[cq][pix] = asum;
  __syncthreads();
  if (cq == 0) {
    float tot = psum[0][pix] + psum[1][pix] + psum[2][pix] + psum[3][pix];
    m[p0 + pix] = tot * (1.0f / 256.0f);
  }
  for (int i = tid; i < 64 * 16; i += 256) {
    int px = i >> 4, ch = i & 15;
    int hw = hw0 + px;
    int h  = hw / Ww;
    int w  = hw - h * Ww;
    *(v4i*)(xb + ((size_t)(n * HP + h + 1) * WP + (w + 1)) * 256 + ch * 16) =
        *(const v4i*)(sb + px * 256 + ch * 16);
  }
}

__global__ void k_boxfilter(const float* __restrict__ m, float* __restrict__ bm) {
  int idx = blockIdx.x * blockDim.x + threadIdx.x;
  if (idx >= NPIX) return;
  int n  = idx / HW;
  int hw = idx - n * HW;
  int h  = hw / Ww;
  int w  = hw - h * Ww;
  const float* mp = m + (size_t)n * HW;
  float s = 0.f;
  #pragma unroll
  for (int dh = -1; dh <= 1; ++dh)
    #pragma unroll
    for (int dw = -1; dw <= 1; ++dw) {
      int hh = h + dh, ww = w + dw;
      if (hh >= 0 && hh < Hh && ww >= 0 && ww < Ww) s += mp[hh * Ww + ww];
    }
  bm[idx] = s * (1.0f / 9.0f);
}

__global__ void k_wprep(const float* __restrict__ W,
                        float* __restrict__ alpha,
                        int8_t* __restrict__ wb) {
  const int co = blockIdx.x;
  const int ci = threadIdx.x;
  const float* wp = W + ((size_t)co * CIN + ci) * 9;
  float s = 0.f;
  #pragma unroll
  for (int t = 0; t < 9; ++t) {
    float v = wp[t];
    s += fabsf(v);
    int8_t sg = (v > 0.f) ? (int8_t)1 : ((v < 0.f) ? (int8_t)-1 : (int8_t)0);
    wb[(size_t)t * 65536 + (ci >> 6) * 16384 + ((ci >> 5) & 1) * 8192 + ((co & 15) + 16 * ((ci >> 3) & 1)) * 256 + (co >> 4) * 16 + ((((ci >> 4) & 1) << 3) | (ci & 7))] = sg;
  }
  __shared__ float ls[256];
  ls[ci] = s;
  __syncthreads();
  for (int off = 128; off > 0; off >>= 1) {
    if (ci < off) ls[ci] += ls[ci + off];
    __syncthreads();
  }
  if (ci == 0) alpha[co] = ls[0] * (1.0f / (float)(CIN * 9));
}

__global__ void k_binconv(const int8_t* __restrict__ xb,
                          const int8_t* __restrict__ wb,
                          const float* __restrict__ bm,
                          const float* __restrict__ alpha,
                          const float* __restrict__ bias,
                          float* __restrict__ out) {
  const int lane = threadIdx.x & 31;
  const int wave = threadIdx.x >> 5;
  const int w0   = blockIdx.x * 16;
  const int nh   = blockIdx.y;
  const int n    = nh / Hh;
  const int h    = nh - n * Hh;
  const int co0  = (blockIdx.z * 8 + wave) * 16;
  const int mrow = lane & 15;
  const int half = lane >> 4;

  const int8_t* abase =
      xb + ((size_t)(n * HP + h) * WP + (w0 + mrow)) * 256 + half * 32;
  const int8_t* bbase = wb + (size_t)(mrow + 16 * half) * 256 + co0;

  v8i acc = {};

  #pragma unroll
  for (int kh = 0; kh < 3; ++kh) {
    #pragma unroll
    for (int kw = 0; kw < 3; ++kw) {
      #pragma unroll
      for (int q = 0; q < 4; ++q) {
        const int aoff = kh * ROWB + kw * 256 + q * 64;
        v4i a0 = *(const v4i*)(abase + aoff);
        v4i a1 = *(const v4i*)(abase + aoff + 16);
        v8i av = __builtin_shufflevector(a0, a1, 0, 1, 2, 3, 4, 5, 6, 7);
        const int boff = (kh * 3 + kw) * 65536 + q * 16384;
        v4i b0 = *(const v4i*)(bbase + boff);
        v4i b1 = *(const v4i*)(bbase + boff + 8192);
        v8i bv = __builtin_shufflevector(b0, b1, 0, 1, 2, 3, 4, 5, 6, 7);
        acc = __builtin_amdgcn_wmma_i32_16x16x64_iu8(
            true, av, true, bv, acc, false, false);
      }
    }
  }

  const int   co  = co0 + mrow;
  const float al  = alpha[co];
  const float bi  = bias[co];
  const float* bmp = bm + (size_t)n * HW + h * Ww;
  float* op = out + ((size_t)(n * COUT + co)) * HW + h * Ww;
  #pragma unroll
  for (int r = 0; r < 8; ++r) {
    int w = w0 + r + half * 8;
    if (w < Ww) {
      float v = ((float)acc[r] + bi) * bmp[w] * al;
      op[w] = v > 0.f ? v : 0.f;
    }
  }
}

extern "C" void kernel_launch(void* const* d_in, const int* in_sizes, int n_in,
                              void* d_out, int out_size, void* d_ws, size_t ws_size,
                              hipStream_t stream) {
  const float* x       = (const float*)d_in[0];
  const float* gamma   = (const float*)d_in[1];
  const float* beta_bn = (const float*)d_in[2];
  const float* W       = (const float*)d_in[3];
  const float* b       = (const float*)d_in[4];
  float* out = (float*)d_out;

  uint8_t* ws = (uint8_t*)d_ws;
  float* scale = (float*)ws;  ws += 256 * sizeof(float);
  float* shift = (float*)ws;  ws += 256 * sizeof(float);
  float* alpha = (float*)ws;  ws += 256 * sizeof(float);
  float* m     = (float*)ws;  ws += (size_t)NPIX * sizeof(float);
  float* bm    = (float*)ws;  ws += (size_t)NPIX * sizeof(float);
  int8_t* wb   = (int8_t*)ws; ws += (size_t)9 * CIN * COUT;
  ws = (uint8_t*)(((uintptr_t)ws + 255) & ~(uintptr_t)255);
  int8_t* xb   = (int8_t*)ws;

  k_bn_stats <<<256, 256, 0, stream>>>(x, gamma, beta_bn, scale, shift);
  k_wprep    <<<256, 256, 0, stream>>>(W, alpha, wb);
  k_padzero  <<<(Nn * 228 * 16 + 255) / 256, 256, 0, stream>>>(xb);
  k_binact   <<<NPIX / 64, 256, 0, stream>>>(x, scale, shift, xb, m);
  k_boxfilter<<<(NPIX + 255) / 256, 256, 0, stream>>>(m, bm);

  dim3 grid(4, Nn * Hh, 2);
  k_binconv  <<<grid, 256, 0, stream>>>(xb, wb, bm, alpha, b, out);
}
